// DLoRF_23587960389934
// MI455X (gfx1250) — matrix operand order repaired and checked by reading, not compiled, not run on hardware
//
#include <hip/hip_runtime.h>

// ---------------------------------------------------------------------------
// DLoRF: out = x @ (U diag(s*mask) V^T)^T  ==  ((x @ U*s) @ V^T)
// Factored path: 137 GFLOP instead of 566 GFLOP.
// bf16 WMMA (v_wmma_f32_16x16x32_bf16), f32 accumulate.
// Wave tile 64x64 (16 WMMA per 16 b128 loads), block tile 128x128 (4 waves).
// ---------------------------------------------------------------------------

typedef __bf16 bf16_t;
typedef __bf16 v16bf __attribute__((ext_vector_type(16)));
typedef __bf16 v8bf  __attribute__((ext_vector_type(8)));
typedef __bf16 v4bf  __attribute__((ext_vector_type(4)));
typedef float  v8f   __attribute__((ext_vector_type(8)));

#define IN_F  4096
#define OUT_F 4096
#define KMAX  512
#define MROWS 16384   // B*S = 8*2048

__device__ __forceinline__ bf16_t f2bf(float f) {
  unsigned u = __builtin_bit_cast(unsigned, f);
  u += 0x7FFFu + ((u >> 16) & 1u);        // round-to-nearest-even
  unsigned short h = (unsigned short)(u >> 16);
  return __builtin_bit_cast(bf16_t, h);
}

// --------------------------- prep kernels ----------------------------------

// float -> bf16, 4 elements per thread (8-byte stores)
__global__ void cvt_f32_bf16_kernel(const float* __restrict__ src,
                                    bf16_t* __restrict__ dst, long long n) {
  long long i = ((long long)blockIdx.x * blockDim.x + threadIdx.x) * 4;
  if (i + 3 < n) {
    float4 f = *(const float4*)(src + i);
    v4bf o;
    o[0] = f2bf(f.x); o[1] = f2bf(f.y); o[2] = f2bf(f.z); o[3] = f2bf(f.w);
    *(v4bf*)(dst + i) = o;
  }
}

// Ust[k][i] = U[i][k] * s[k] * mask[k]   (U:[IN_F,KMAX] -> Ust:[KMAX,IN_F] bf16)
__global__ void build_ust_kernel(const float* __restrict__ U,
                                 const float* __restrict__ s,
                                 const unsigned char* __restrict__ mask,
                                 bf16_t* __restrict__ ust) {
  int i = blockIdx.x * blockDim.x + threadIdx.x;   // 0..IN_F-1
  int k = blockIdx.y;                              // 0..KMAX-1
  float sm = s[k] * (mask[k] ? 1.0f : 0.0f);
  ust[(size_t)k * IN_F + i] = f2bf(U[(size_t)i * KMAX + k] * sm);
}

// --------------------------- WMMA GEMM -------------------------------------
// C[M,N] = A[M,K] @ B[N,K]^T, A/B bf16 row-major, f32 accumulate.
// Block: 128 threads = 4 waves in 2(M) x 2(N); block tile 128 x 128.
// Wave tile 64x64: 4 A-frags x 4 B-frags -> 16 WMMA per K-step of 32.
// Writes f32 (Cf) or bf16 (Cb) depending on which pointer is non-null.

#define BM 128
#define BN 128

__global__ __launch_bounds__(128)
void gemm_nt_bf16_kernel(const bf16_t* __restrict__ A,
                         const bf16_t* __restrict__ B,
                         float*  __restrict__ Cf,
                         bf16_t* __restrict__ Cb,
                         int N, int K) {
  const int lane    = threadIdx.x & 31;
  const int wave    = threadIdx.x >> 5;
  const int wave_m  = wave >> 1;        // 0..1
  const int wave_n  = wave & 1;         // 0..1
  const int lr      = lane & 15;        // row/col within 16-wide group
  const int hsel    = lane >> 4;        // 0: lanes 0-15, 1: lanes 16-31
  const int m0      = blockIdx.x * BM + wave_m * 64;
  const int n0      = blockIdx.y * BN + wave_n * 64;

  // A fragment base: lane lr holds row m, K-halves {0..7,16..23} (hsel=0)
  // or {8..15,24..31} (hsel=1)  [ISA 7.12.2, 16-bit A 16x32]
  const bf16_t* a_ptr  = A + (size_t)(m0 + lr) * K + (hsel ? 8 : 0);
  const size_t  a_tile = (size_t)16 * K;              // stride between M tiles
  // B fragment base: lane lr holds column n, K-halves {0..7,16..23} (hsel=0)
  // or {8..15,24..31} (hsel=1), same as A
  const bf16_t* b_ptr  = B + (size_t)(n0 + lr) * K + (hsel ? 8 : 0);
  const size_t  b_tile = (size_t)16 * K;              // stride between N tiles

  v8f c[4][4];
#pragma unroll
  for (int i = 0; i < 4; ++i)
#pragma unroll
    for (int j = 0; j < 4; ++j)
#pragma unroll
      for (int r = 0; r < 8; ++r) c[i][j][r] = 0.0f;

  for (int k0 = 0; k0 < K; k0 += 32) {
    // A fragments (two 16-byte loads each, ISA half-layout)
    v16bf a[4];
#pragma unroll
    for (int i = 0; i < 4; ++i) {
      v8bf alo = *(const v8bf*)(a_ptr + i * a_tile + k0);
      v8bf ahi = *(const v8bf*)(a_ptr + i * a_tile + k0 + 16);
      a[i] = __builtin_shufflevector(alo, ahi,
          0,1,2,3,4,5,6,7,8,9,10,11,12,13,14,15);
    }
    // B fragments: two 16-byte loads each, same half-layout as A
    v16bf b[4];
#pragma unroll
    for (int j = 0; j < 4; ++j) {
      v8bf blo = *(const v8bf*)(b_ptr + j * b_tile + k0);
      v8bf bhi = *(const v8bf*)(b_ptr + j * b_tile + k0 + 16);
      b[j] = __builtin_shufflevector(blo, bhi,
          0,1,2,3,4,5,6,7,8,9,10,11,12,13,14,15);
    }

#pragma unroll
    for (int i = 0; i < 4; ++i)
#pragma unroll
      for (int j = 0; j < 4; ++j)
        c[i][j] = __builtin_amdgcn_wmma_f32_16x16x32_bf16(
            false, a[i], false, b[j], (short)0, c[i][j], false, false);
  }

  // Store: C/D layout — VGPR r = row (m0 + i*16 + hsel*8 + r), col n0+j*16+lr
#pragma unroll
  for (int i = 0; i < 4; ++i) {
    const int rbase = m0 + i * 16 + hsel * 8;
#pragma unroll
    for (int j = 0; j < 4; ++j) {
      const int col = n0 + j * 16 + lr;
      if (Cf) {
#pragma unroll
        for (int r = 0; r < 8; ++r)
          Cf[(size_t)(rbase + r) * N + col] = c[i][j][r];
      } else {
#pragma unroll
        for (int r = 0; r < 8; ++r)
          Cb[(size_t)(rbase + r) * N + col] = f2bf(c[i][j][r]);
      }
    }
  }
}

// --------------------------- launcher --------------------------------------

extern "C" void kernel_launch(void* const* d_in, const int* in_sizes, int n_in,
                              void* d_out, int out_size, void* d_ws, size_t ws_size,
                              hipStream_t stream) {
  const float*         x    = (const float*)d_in[0];   // [16384, 4096]
  const float*         U    = (const float*)d_in[1];   // [4096, 512]
  const float*         V    = (const float*)d_in[2];   // [4096, 512]
  const float*         s    = (const float*)d_in[3];   // [512]
  const unsigned char* mask = (const unsigned char*)d_in[4]; // [512] bool
  float*               out  = (float*)d_out;           // [16384, 4096]

  // workspace layout (all 256B aligned)
  char* ws = (char*)d_ws;
  const size_t XB_BYTES  = (size_t)MROWS * IN_F * sizeof(bf16_t);   // 128 MB
  const size_t UST_BYTES = (size_t)KMAX * IN_F * sizeof(bf16_t);    //   4 MB
  const size_t VB_BYTES  = (size_t)OUT_F * KMAX * sizeof(bf16_t);   //   4 MB
  bf16_t* xb  = (bf16_t*)(ws);
  bf16_t* ust = (bf16_t*)(ws + XB_BYTES);
  bf16_t* vb  = (bf16_t*)(ws + XB_BYTES + UST_BYTES);
  bf16_t* T   = (bf16_t*)(ws + XB_BYTES + UST_BYTES + VB_BYTES);    //  16 MB

  // 1) conversions
  {
    long long n = (long long)MROWS * IN_F;             // 67,108,864
    cvt_f32_bf16_kernel<<<(unsigned)(n / 4 / 256), 256, 0, stream>>>(x, xb, n);
  }
  build_ust_kernel<<<dim3(IN_F / 256, KMAX), 256, 0, stream>>>(U, s, mask, ust);
  {
    long long n = (long long)OUT_F * KMAX;             // 2,097,152
    cvt_f32_bf16_kernel<<<(unsigned)(n / 4 / 256), 256, 0, stream>>>(V, vb, n);
  }

  // 2) GEMM1: T[16384,512] = xb[16384,4096] @ Ust[512,4096]^T   (bf16 out)
  gemm_nt_bf16_kernel<<<dim3(MROWS / BM, KMAX / BN), 128, 0, stream>>>(
      xb, ust, nullptr, T, KMAX, IN_F);

  // 3) GEMM2: out[16384,4096] = T[16384,512] @ Vb[4096,512]^T   (f32 out)
  gemm_nt_bf16_kernel<<<dim3(MROWS / BM, OUT_F / BN), 128, 0, stream>>>(
      T, vb, out, nullptr, OUT_F, KMAX);
}
